// HybridQLSTM_65481071398288
// MI455X (gfx1250) — matrix operand order repaired and checked by reading, not compiled, not run on hardware
//
#include <hip/hip_runtime.h>

// ---------------------------------------------------------------------------
// LSTM on MI455X (gfx1250): fp16 WMMA GEMM per step, fused gates + state
// update. Triple-buffered LDS weight staging via gfx1250
// GLOBAL_LOAD_ASYNC_TO_LDS_B128 (count-based s_wait_asynccnt so staging stays
// in flight across the whole iteration), register-prefetched A fragments.
// ---------------------------------------------------------------------------

typedef __attribute__((ext_vector_type(16))) _Float16 v16h;
typedef __attribute__((ext_vector_type(8)))  _Float16 v8h;
typedef __attribute__((ext_vector_type(8)))  float    v8f;
typedef int v4i_gcc __attribute__((vector_size(16)));   // matches builtin proto V4i

#define T_STEPS 256
#define BATCH   256
#define DIN     512
#define HID     1024
#define KDIM    (DIN + HID)     // 1536
#define NTOT    (4 * HID)       // 4096 fused gate columns
#define KSTAGE  64              // K depth staged per barrier (2 WMMA K-steps)
#define NITER   (KDIM / KSTAGE) // 24 pipeline iterations

#if __has_builtin(__builtin_amdgcn_global_load_async_to_lds_b128) && \
    __has_builtin(__builtin_amdgcn_s_wait_asynccnt)
#define HAVE_ASYNC_LDS 1
#else
#define HAVE_ASYNC_LDS 0
#endif

union V16 { v16h v; v8h h[2]; };

__device__ __forceinline__ float sigm_f(float x) {
    return 1.0f / (1.0f + __expf(-x));
}
__device__ __forceinline__ float tanh_f(float x) {
    return 1.0f - 2.0f / (__expf(2.0f * x) + 1.0f);
}

// Pack [Wf|Wi|Wg|Wo] (each [KDIM,HID] fp32) into transposed fp16 Wt[n][k].
__global__ void pack_weights(const float* __restrict__ Wf, const float* __restrict__ Wi,
                             const float* __restrict__ Wg, const float* __restrict__ Wo,
                             _Float16* __restrict__ Wt) {
    size_t idx = (size_t)blockIdx.x * blockDim.x + threadIdx.x;
    if (idx >= (size_t)NTOT * KDIM) return;
    int n = (int)(idx / KDIM);
    int k = (int)(idx % KDIM);
    int g = n >> 10;
    int col = n & (HID - 1);
    const float* W = (g == 0) ? Wf : (g == 1) ? Wi : (g == 2) ? Wg : Wo;
    Wt[idx] = (_Float16)W[(size_t)k * HID + col];
}

__global__ void init_state(float* __restrict__ c, _Float16* __restrict__ combined0) {
    int idx = blockIdx.x * blockDim.x + threadIdx.x;
    if (idx >= BATCH * HID) return;
    c[idx] = 0.0f;
    int bm = idx / HID, n = idx % HID;
    combined0[(size_t)bm * KDIM + DIN + n] = (_Float16)0.0f;
}

__global__ void convert_x(const float* __restrict__ x, _Float16* __restrict__ combined) {
    int idx = blockIdx.x * blockDim.x + threadIdx.x;
    if (idx >= BATCH * DIN) return;
    int bm = idx / DIN, d = idx % DIN;
    combined[(size_t)bm * KDIM + d] = (_Float16)x[idx];
}

// One LSTM step. Grid (HID/16, BATCH/128), 256 threads = 8 waves.
// Wave = one 16(batch)x16(hidden) tile, 4 gate accumulators sharing A.
__global__ __launch_bounds__(256)
void lstm_step(const _Float16* __restrict__ combined,   // [BATCH][KDIM] fp16 (x_t | h_{t-1})
               const _Float16* __restrict__ Wt,         // [NTOT][KDIM] fp16 transposed weights
               const float* __restrict__ bfp, const float* __restrict__ bip,
               const float* __restrict__ bgp, const float* __restrict__ bop,
               float* __restrict__ c, float* __restrict__ out_t,
               _Float16* __restrict__ hdst)
{
    __shared__ __align__(16) _Float16 ldsB[3][64 * KSTAGE];   // 3 x 8 KB rotation

    const int tid     = threadIdx.x;
    const int wave    = tid >> 5;
    const int lane    = tid & 31;
    const int h0      = blockIdx.x * 16;
    const int m0      = blockIdx.y * 128 + wave * 16;
    const int halfsel = (lane < 16) ? 0 : 1;
    const int nl      = lane & 15;

    v8f acc[4] = {};

    // staging geometry: 64 rows (gate*16+n) x 128B; each thread moves 32B/stage
    const int srow   = tid >> 2;            // 0..63
    const int schunk = (tid & 3) * 16;      // half-offset within row: 0,16,32,48
    const int sgate  = srow >> 4;
    const int sn     = srow & 15;
    const _Float16* gsrc_row = Wt + (size_t)(sgate * HID + h0 + sn) * KDIM + schunk;
    _Float16* ldst[3] = { &ldsB[0][srow * KSTAGE + schunk],
                          &ldsB[1][srow * KSTAGE + schunk],
                          &ldsB[2][srow * KSTAGE + schunk] };

    const _Float16* abase = combined + (size_t)(m0 + nl) * KDIM + halfsel * 8;

#if HAVE_ASYNC_LDS
    // -------- async global->LDS pipeline (ASYNCcnt, count-based waits) ------
    // builtin proto: (v4i AS1* gsrc, v4i AS3* ldst, imm offset, imm cpol)
    #define STAGE_ASYNC(kk, dst) do {                                          \
        _Float16* _g = (_Float16*)(gsrc_row + (kk));                           \
        _Float16* _d = (_Float16*)(dst);                                       \
        __builtin_amdgcn_global_load_async_to_lds_b128(                        \
            (__attribute__((address_space(1))) v4i_gcc*)(_g),                  \
            (__attribute__((address_space(3))) v4i_gcc*)(_d), 0, 0);           \
        __builtin_amdgcn_global_load_async_to_lds_b128(                        \
            (__attribute__((address_space(1))) v4i_gcc*)(_g + 8),              \
            (__attribute__((address_space(3))) v4i_gcc*)(_d + 8), 0, 0);       \
    } while (0)

    STAGE_ASYNC(0, ldst[0]);
    V16 a0, a1;                              // A fragments for current iteration
    a0.h[0] = *(const v8h*)(abase + 0);
    a0.h[1] = *(const v8h*)(abase + 16);
    a1.h[0] = *(const v8h*)(abase + 32);
    a1.h[1] = *(const v8h*)(abase + 48);

    #pragma unroll 3
    for (int i = 0; i < NITER; ++i) {
        const int nkk = (i + 1 < NITER) ? (i + 1) * KSTAGE : 0; // dummy at tail
        // Issue next stage FIRST, then wait only for the *older* stage (<=2
        // outstanding). Async loads retire in order, so stage(i) is in LDS
        // while stage(i+1) stays in flight across this whole iteration.
        STAGE_ASYNC(nkk, ldst[(i + 1) % 3]);
        __builtin_amdgcn_s_wait_asynccnt(2);
        __syncthreads();                        // everyone's stage(i) visible
        // prefetch next iteration's A fragments (consumed next iter)
        V16 na0, na1;
        na0.h[0] = *(const v8h*)(abase + nkk);
        na0.h[1] = *(const v8h*)(abase + nkk + 16);
        na1.h[0] = *(const v8h*)(abase + nkk + 32);
        na1.h[1] = *(const v8h*)(abase + nkk + 48);

        const _Float16* lbuf = ldsB[i % 3];
        #pragma unroll
        for (int ks = 0; ks < 2; ++ks) {
            const V16& af = ks ? a1 : a0;
            V16 bfr[4];
            #pragma unroll
            for (int g = 0; g < 4; ++g) {
                const _Float16* bb = &lbuf[(g * 16 + nl) * KSTAGE + ks * 32 + halfsel * 8];
                bfr[g].h[0] = *(const v8h*)(bb);
                bfr[g].h[1] = *(const v8h*)(bb + 16);
            }
            #pragma unroll
            for (int g = 0; g < 4; ++g)
                acc[g] = __builtin_amdgcn_wmma_f32_16x16x32_f16(
                    false, af.v, false, bfr[g].v, (short)0, acc[g], false, false);
        }
        a0 = na0;
        a1 = na1;
    }
    __builtin_amdgcn_s_wait_asynccnt(0);
#else
    // -------- register-staged fallback pipeline --------
    struct Stage { v8h lo, hi; };
    Stage sreg;
    sreg.lo = *(const v8h*)(gsrc_row + 0);
    sreg.hi = *(const v8h*)(gsrc_row + 8);
    *(v8h*)(ldst[0] + 0) = sreg.lo;
    *(v8h*)(ldst[0] + 8) = sreg.hi;
    sreg.lo = *(const v8h*)(gsrc_row + KSTAGE + 0);
    sreg.hi = *(const v8h*)(gsrc_row + KSTAGE + 8);

    #pragma unroll 2
    for (int i = 0; i < NITER; ++i) {
        const int kk  = i * KSTAGE;
        const int nkk = (i + 2 < NITER) ? (i + 2) * KSTAGE : 0;
        __syncthreads();                        // publishes stores to buf(i&1)
        *(v8h*)(ldst[(i + 1) & 1] + 0) = sreg.lo;
        *(v8h*)(ldst[(i + 1) & 1] + 8) = sreg.hi;
        sreg.lo = *(const v8h*)(gsrc_row + nkk + 0);
        sreg.hi = *(const v8h*)(gsrc_row + nkk + 8);
        const _Float16* lbuf = ldsB[i & 1];
        #pragma unroll
        for (int ks = 0; ks < 2; ++ks) {
            V16 afrag;
            afrag.h[0] = *(const v8h*)(abase + kk + ks * 32);
            afrag.h[1] = *(const v8h*)(abase + kk + ks * 32 + 16);
            V16 bfr[4];
            #pragma unroll
            for (int g = 0; g < 4; ++g) {
                const _Float16* bb = &lbuf[(g * 16 + nl) * KSTAGE + ks * 32 + halfsel * 8];
                bfr[g].h[0] = *(const v8h*)(bb);
                bfr[g].h[1] = *(const v8h*)(bb + 16);
            }
            #pragma unroll
            for (int g = 0; g < 4; ++g)
                acc[g] = __builtin_amdgcn_wmma_f32_16x16x32_f16(
                    false, afrag.v, false, bfr[g].v, (short)0, acc[g], false, false);
        }
    }
#endif

    // Pointwise epilogue; C/D layout -> consecutive lanes = consecutive columns
    const int   ncol = h0 + nl;
    const float bfv = bfp[ncol], biv = bip[ncol], bgv = bgp[ncol], bov = bop[ncol];

    #pragma unroll
    for (int r = 0; r < 8; ++r) {
        const int m = m0 + r + halfsel * 8;
        const float fg = sigm_f(acc[0][r] + bfv);
        const float ig = sigm_f(acc[1][r] + biv);
        const float gg = tanh_f(acc[2][r] + bgv);
        const float og = sigm_f(acc[3][r] + bov);
        const size_t ci = (size_t)m * HID + ncol;
        const float cn = fg * c[ci] + ig * gg;
        c[ci] = cn;
        const float hv = og * tanh_f(cn);
        out_t[ci] = hv;
        hdst[(size_t)m * KDIM + ncol] = (_Float16)hv;
    }
}

__global__ void finalize(const float* __restrict__ last_h, const float* __restrict__ c,
                         float* __restrict__ hx, float* __restrict__ cx) {
    int idx = blockIdx.x * blockDim.x + threadIdx.x;
    if (idx >= BATCH * HID) return;
    hx[idx] = last_h[idx];
    cx[idx] = c[idx];
}

extern "C" void kernel_launch(void* const* d_in, const int* in_sizes, int n_in,
                              void* d_out, int out_size, void* d_ws, size_t ws_size,
                              hipStream_t stream) {
    const float* inputs = (const float*)d_in[0];
    const float* Wf = (const float*)d_in[1];
    const float* bf = (const float*)d_in[2];
    const float* Wi = (const float*)d_in[3];
    const float* bi = (const float*)d_in[4];
    const float* Wg = (const float*)d_in[5];
    const float* bg = (const float*)d_in[6];
    const float* Wo = (const float*)d_in[7];
    const float* bo = (const float*)d_in[8];
    float* out = (float*)d_out;

    char* ws = (char*)d_ws;
    size_t off = 0;
    _Float16* Wt = (_Float16*)(ws + off);
    off += (size_t)NTOT * KDIM * sizeof(_Float16);  off = (off + 255) & ~(size_t)255;
    _Float16* combA = (_Float16*)(ws + off);
    off += (size_t)BATCH * KDIM * sizeof(_Float16); off = (off + 255) & ~(size_t)255;
    _Float16* combB = (_Float16*)(ws + off);
    off += (size_t)BATCH * KDIM * sizeof(_Float16); off = (off + 255) & ~(size_t)255;
    float* c = (float*)(ws + off);

    pack_weights<<<((size_t)NTOT * KDIM + 255) / 256, 256, 0, stream>>>(Wf, Wi, Wg, Wo, Wt);
    init_state<<<(BATCH * HID) / 256, 256, 0, stream>>>(c, combA);

    for (int t = 0; t < T_STEPS; ++t) {
        _Float16* cur = (t & 1) ? combB : combA;
        _Float16* nxt = (t & 1) ? combA : combB;
        convert_x<<<(BATCH * DIN) / 256, 256, 0, stream>>>(
            inputs + (size_t)t * BATCH * DIN, cur);
        lstm_step<<<dim3(HID / 16, BATCH / 128), 256, 0, stream>>>(
            cur, Wt, bf, bi, bg, bo, c,
            out + (size_t)t * BATCH * HID, nxt + DIN);
    }

    float* hx = out + (size_t)T_STEPS * BATCH * HID;
    float* cx = hx + (size_t)BATCH * HID;
    finalize<<<(BATCH * HID) / 256, 256, 0, stream>>>(
        out + (size_t)(T_STEPS - 1) * BATCH * HID, c, hx, cx);
}
